// ACCPC_MA_8383776162157
// MI455X (gfx1250) — compile-verified, matrix operand order checked by reading, not run on hardware
//
#include <hip/hip_runtime.h>

typedef __attribute__((ext_vector_type(16))) _Float16 v16h;
typedef __attribute__((ext_vector_type(8)))  _Float16 v8h;
typedef __attribute__((ext_vector_type(8)))  float    v8f;
typedef __attribute__((ext_vector_type(4)))  unsigned int u32x4;
typedef __attribute__((ext_vector_type(8)))  unsigned int u32x8;

// ---------------------------------------------------------------------------
// WMMA helper (CDNA5 gfx1250, wave32): D = A(16x32 f16) x B(32x16 f16) + C(f32)
// ---------------------------------------------------------------------------
__device__ __forceinline__ v8f wmma_f16(v16h a, v16h b, v8f c) {
  return __builtin_amdgcn_wmma_f32_16x16x32_f16(false, a, false, b, (short)0, c,
                                                false, false);
}

// Load a 16x32 f16 fragment from an LDS tile (row-major, rowStride halves).
// koff = starting K offset (halves, multiple of 32). Layout per ISA 7.12.2.
__device__ __forceinline__ v16h fragS(const _Float16* tile, int row0, int rowStride,
                                      int koff) {
  const int lane = threadIdx.x & 31;
  const v8h* p =
      (const v8h*)(tile + (size_t)(row0 + (lane & 15)) * rowStride + koff);
  const int sel = lane >> 4;
  v8h lo = p[sel];
  v8h hi = p[2 + sel];
  v16h r;
#pragma unroll
  for (int i = 0; i < 8; ++i) { r[i] = lo[i]; r[i + 8] = hi[i]; }
  return r;
}

// Per-lane async global->LDS copies (CDNA5): nbytes in {32, 64} per call site.
__device__ __forceinline__ void async_copy_b128x2(unsigned int la,
                                                  unsigned long long ga) {
  asm volatile("global_load_async_to_lds_b128 %0, %1, off\n\t"
               "global_load_async_to_lds_b128 %0, %1, off offset:16"
               :: "v"(la), "v"(ga) : "memory");
}
__device__ __forceinline__ void async_copy_b128x4(unsigned int la,
                                                  unsigned long long ga) {
  asm volatile("global_load_async_to_lds_b128 %0, %1, off\n\t"
               "global_load_async_to_lds_b128 %0, %1, off offset:16\n\t"
               "global_load_async_to_lds_b128 %0, %1, off offset:32\n\t"
               "global_load_async_to_lds_b128 %0, %1, off offset:48"
               :: "v"(la), "v"(ga) : "memory");
}
__device__ __forceinline__ void wait_async0() {
  asm volatile("s_wait_asynccnt 0x0" ::: "memory");
}

// ---------------------------------------------------------------------------
// One-time operand staging kernels
// ---------------------------------------------------------------------------
__global__ void cvt_f32_f16(const float* __restrict__ in, _Float16* __restrict__ out,
                            int n) {
  const int i = (blockIdx.x * 256 + threadIdx.x) * 8;
  if (i >= n) return;
  v8h o;
#pragma unroll
  for (int j = 0; j < 8; ++j) o[j] = (_Float16)in[i + j];
  *(v8h*)(out + i) = o;
}

// W(K x 512, f32, row-major) -> Wt(512 x K, f16, N-major). For W1 reads only
// the first 512 rows (the proj part).
__global__ void transpose_w(const float* __restrict__ W, _Float16* __restrict__ Wt,
                            int Kdim) {
  const int e = (blockIdx.x * 256 + threadIdx.x) * 8;   // over 512*Kdim
  const int n = e / Kdim, kq = e % Kdim;
  v8h o;
#pragma unroll
  for (int j = 0; j < 8; ++j) o[j] = (_Float16)W[(size_t)(kq + j) * 512 + n];
  *(v8h*)(Wt + e) = o;
}

// ---------------------------------------------------------------------------
// GEMM tile geometry: block tile 64 rows x 128 cols, 256 threads (8 wave32).
// Wave w: row band rb = w&3, col half ch = w>>2 (4 WMMA tiles). K-chunk 64.
// All tile loads are CDNA5 async global->LDS (no VGPR staging, no ds_store).
// ---------------------------------------------------------------------------
__global__ void gemm_plain(const _Float16* __restrict__ A, const _Float16* __restrict__ Wt,
                           const float* __restrict__ bias, _Float16* __restrict__ out,
                           int Kdim) {
  __shared__ __attribute__((aligned(16))) _Float16 As[64][64];
  __shared__ __attribute__((aligned(16))) _Float16 Bt[128][64];
  const int tid = threadIdx.x, lane = tid & 31, wave = tid >> 5;
  const int rb = wave & 3, ch = wave >> 2;
  const int row0 = blockIdx.x * 64, col0 = blockIdx.y * 128;
  v8f acc[4] = {};
  for (int kk = 0; kk < Kdim; kk += 64) {
    { // A chunk 64x64: 2 x b128 async per thread
      const int e = tid * 16, r = e >> 6, c = e & 63;
      const _Float16* src = A + (size_t)(row0 + r) * Kdim + kk + c;
      __builtin_prefetch(src + 64, 0, 1);
      async_copy_b128x2((unsigned int)(uintptr_t)&As[r][c],
                        (unsigned long long)(uintptr_t)src);
    }
    { // B chunk 128x64: 4 x b128 async per thread (contiguous in Wt)
      const int e = tid * 32, n = e >> 6, c = e & 63;
      const _Float16* src = Wt + (size_t)(col0 + n) * Kdim + kk + c;
      async_copy_b128x4((unsigned int)(uintptr_t)&Bt[n][c],
                        (unsigned long long)(uintptr_t)src);
    }
    wait_async0();
    __syncthreads();
#pragma unroll
    for (int s = 0; s < 2; ++s) {
      v16h af = fragS(&As[0][0], rb * 16, 64, s * 32);
#pragma unroll
      for (int j = 0; j < 4; ++j) {
        v16h bf = fragS(&Bt[0][0], ch * 64 + j * 16, 64, s * 32);
        acc[j] = wmma_f16(af, bf, acc[j]);
      }
    }
    __syncthreads();
  }
  const int hi8 = (lane >> 4) * 8;
  const int cb0 = col0 + ch * 64 + (lane & 15);
  float bs[4];
#pragma unroll
  for (int j = 0; j < 4; ++j) bs[j] = bias[cb0 + j * 16];
#pragma unroll
  for (int v = 0; v < 8; ++v) {
    const size_t r = (size_t)(row0 + rb * 16 + hi8 + v);
#pragma unroll
    for (int j = 0; j < 4; ++j)
      out[r * 512 + cb0 + j * 16] = (_Float16)(acc[j][v] + bs[j]);
  }
}

// GEMM (hidden): h = relu(proj_rows @ W1T + win @ W1f32[512:512+6k] + b1).
// Row r -> (a = r/(tk*128), t = (r/128)%tk, b = r%128); proj row (a*32+t)*128+b.
__global__ void gemm_h(const _Float16* __restrict__ proj, const _Float16* __restrict__ W1T,
                       const float* __restrict__ W1f, const float* __restrict__ b1,
                       const float* __restrict__ actions, _Float16* __restrict__ hout,
                       int tk, int k) {
  __shared__ __attribute__((aligned(16))) _Float16 As[64][64];
  __shared__ __attribute__((aligned(16))) _Float16 Bt[128][64];
  const int tid = threadIdx.x, lane = tid & 31, wave = tid >> 5;
  const int rb = wave & 3, ch = wave >> 2;
  const int row0 = blockIdx.x * 64, col0 = blockIdx.y * 128;
  const int span = tk * 128;
  v8f acc[4] = {};
  for (int kk = 0; kk < 512; kk += 64) {
    {
      const int e = tid * 16, r = e >> 6, c = e & 63;
      const int rg = row0 + r;
      const int a = rg / span, rem = rg - a * span;
      const int t = rem >> 7, b = rem & 127;
      const _Float16* src =
          proj + (((size_t)(a * 32 + t)) * 128 + b) * 512 + kk + c;
      async_copy_b128x2((unsigned int)(uintptr_t)&As[r][c],
                        (unsigned long long)(uintptr_t)src);
    }
    {
      const int e = tid * 32, n = e >> 6, c = e & 63;
      const _Float16* src = W1T + (size_t)(col0 + n) * 512 + kk + c;
      async_copy_b128x4((unsigned int)(uintptr_t)&Bt[n][c],
                        (unsigned long long)(uintptr_t)src);
    }
    wait_async0();
    __syncthreads();
#pragma unroll
    for (int s = 0; s < 2; ++s) {
      v16h af = fragS(&As[0][0], rb * 16, 64, s * 32);
#pragma unroll
      for (int j = 0; j < 4; ++j) {
        v16h bf = fragS(&Bt[0][0], ch * 64 + j * 16, 64, s * 32);
        acc[j] = wmma_f16(af, bf, acc[j]);
      }
    }
    __syncthreads();
  }
  const int hi8 = (lane >> 4) * 8;
  const int cb0 = col0 + ch * 64 + (lane & 15);
  float bb[4];
#pragma unroll
  for (int j = 0; j < 4; ++j) bb[j] = b1[cb0 + j * 16];
#pragma unroll
  for (int v = 0; v < 8; ++v) {
    const int rg = row0 + rb * 16 + hi8 + v;
    const int a = rg / span, rem = rg - a * span;
    const int t = rem >> 7, b = rem & 127;
    float s[4] = {bb[0], bb[1], bb[2], bb[3]};
    for (int dt = 0; dt < k; ++dt) {   // action-window term, inner dim k*6 <= 18
      const float* arow = actions + (((size_t)b * 8 + a) * 32 + (t + dt)) * 6;
      for (int j6 = 0; j6 < 6; ++j6) {
        const float av = arow[j6];
        const float* wrow = W1f + (size_t)(512 + dt * 6 + j6) * 512;
#pragma unroll
        for (int j = 0; j < 4; ++j) s[j] += av * wrow[cb0 + j * 16];
      }
    }
    const size_t r = (size_t)rg;
#pragma unroll
    for (int j = 0; j < 4; ++j)
      hout[r * 512 + cb0 + j * 16] = (_Float16)fmaxf(acc[j][v] + s[j], 0.f);
  }
}

// ---------------------------------------------------------------------------
// Row-wise L2 normalize (in place), f16 rows of length 512; one wave per row
// ---------------------------------------------------------------------------
__global__ void normalize_rows(_Float16* __restrict__ buf, int rows) {
  const int row = blockIdx.x * 8 + (threadIdx.x >> 5);
  const int lane = threadIdx.x & 31;
  _Float16* p = buf + (size_t)row * 512;
  float ss = 0.f;
  float vals[16];
#pragma unroll
  for (int i = 0; i < 16; ++i) { vals[i] = (float)p[lane + i * 32]; ss += vals[i] * vals[i]; }
#pragma unroll
  for (int m = 16; m >= 1; m >>= 1) ss += __shfl_xor(ss, m, 32);
  const float inv = 1.0f / fmaxf(sqrtf(ss), 1e-12f);
#pragma unroll
  for (int i = 0; i < 16; ++i) p[lane + i * 32] = (_Float16)(vals[i] * inv);
}

// ---------------------------------------------------------------------------
// Stage D: logits + cross-entropy reduction.
// Dynamic LDS: Pf[64][512] (64KB, loaded ONCE via async), Qt[128][128] (32KB,
// streamed by the Tensor Data Mover), S[64][128] (32KB), red/rowsum (~1.3KB).
// Per c-block: 4 TDM chunks x 16 WMMAs, then per-row LSE - diag, mask c != a.
// One deterministic partial per workgroup.
// ---------------------------------------------------------------------------
__global__ void logits_ce(const _Float16* __restrict__ pred, const _Float16* __restrict__ tgt,
                          float* __restrict__ partials, int tk, int k, int pbase) {
  extern __shared__ __attribute__((aligned(16))) char smem[];
  _Float16* Pf = (_Float16*)smem;                   // [64][512]
  _Float16* Qt = (_Float16*)(smem + 65536);         // [128][128]
  float* S      = (float*)(smem + 98304);           // [64][128]
  float* red    = (float*)(smem + 131072);          // [64][4]
  float* rowsum = (float*)(smem + 132096);          // [64]
  const int t = blockIdx.x, blk = blockIdx.y;
  const int tid = threadIdx.x, lane = tid & 31, wave = tid >> 5;
  const int rb = wave & 3, ch = wave >> 2;
  const int row0 = blk * 64;
  const int row = tid >> 2, q = tid & 3;   // LSE reduction: 4 threads per row
  float rowacc = 0.f;                      // meaningful for q == 0 threads

  { // load all 64 P rows once: 128 halves (256B) per thread, async->LDS
    const int r = tid >> 2, c = (tid & 3) << 7;
    const int rg = row0 + r, a = rg >> 7, b = rg & 127;
    const _Float16* src = pred + (((size_t)(a * tk + t)) * 128 + b) * 512 + c;
    const unsigned long long ga = (unsigned long long)(uintptr_t)src;
    const unsigned int la = (unsigned int)(uintptr_t)(Pf + (size_t)r * 512 + c);
#pragma unroll
    for (int g = 0; g < 4; ++g)
      async_copy_b128x4(la + g * 64, ga + g * 64);
    wait_async0();
  }
  __syncthreads();

  for (int cb = 0; cb < 8; ++cb) {
    v8f acc[4] = {};
    for (int kk = 0; kk < 512; kk += 128) {
      if (wave == 0) {
        // Tensor Data Mover: 2D tile, 128 rows x 128 f16, row stride 512 elems
        const _Float16* qsrc =
            tgt + (((size_t)(cb * 35 + t + k)) * 128) * 512 + kk;
        const unsigned long long ga = (unsigned long long)(uintptr_t)qsrc;
        const unsigned int la = (unsigned int)(uintptr_t)Qt;
        u32x4 g0;
        g0[0] = 1u;                                   // count=1, user mode
        g0[1] = la;                                   // lds_addr
        g0[2] = (unsigned int)ga;                     // global_addr[31:0]
        g0[3] = (unsigned int)((ga >> 32) & 0x1FFFFFFull) | (2u << 30); // [56:32], type=2
        u32x8 g1;
        g1[0] = 1u << 16;                             // data_size = 2 bytes
        g1[1] = 128u << 16;                           // tensor_dim0 = 128 (lo16)
        g1[2] = 128u << 16;                           // tdim0 hi=0 | tensor_dim1 = 128
        g1[3] = 128u << 16;                           // tdim1 hi=0 | tile_dim0 = 128
        g1[4] = 128u;                                 // tile_dim1 = 128, tile_dim2 = 0
        g1[5] = 512u;                                 // tensor_dim0_stride = 512
        g1[6] = 0u;
        g1[7] = 0u;
        asm volatile("tensor_load_to_lds %0, %1" :: "s"(g0), "s"(g1) : "memory");
        __builtin_amdgcn_s_wait_tensorcnt(0);
      }
      __syncthreads();
#pragma unroll
      for (int s = 0; s < 4; ++s) {
        v16h af = fragS(Pf, rb * 16, 512, kk + s * 32);
#pragma unroll
        for (int j = 0; j < 4; ++j) {
          v16h bf = fragS(Qt, ch * 64 + j * 16, 128, s * 32);
          acc[j] = wmma_f16(af, bf, acc[j]);
        }
      }
      __syncthreads();
    }
    // write scaled logits block to LDS
    const int hi8 = (lane >> 4) * 8;
#pragma unroll
    for (int j = 0; j < 4; ++j) {
      const int cc = ch * 64 + j * 16 + (lane & 15);
#pragma unroll
      for (int v = 0; v < 8; ++v)
        S[(rb * 16 + hi8 + v) * 128 + cc] = acc[j][v] * 10.0f;   // 1/TEMP
    }
    __syncthreads();
    // per-row LSE over 128 cols (4 threads per row, 32 cols each)
    const float* srow = &S[row * 128 + q * 32];
    float m = -1e30f;
    for (int i = 0; i < 32; ++i) m = fmaxf(m, srow[i]);
    red[row * 4 + q] = m;
    __syncthreads();
    const float rm = fmaxf(fmaxf(red[row * 4 + 0], red[row * 4 + 1]),
                           fmaxf(red[row * 4 + 2], red[row * 4 + 3]));
    __syncthreads();
    float s = 0.f;
    for (int i = 0; i < 32; ++i) s += __expf(srow[i] - rm);
    red[row * 4 + q] = s;
    __syncthreads();
    if (q == 0) {
      const float ssum = red[row * 4 + 0] + red[row * 4 + 1] +
                         red[row * 4 + 2] + red[row * 4 + 3];
      const float lse = rm + __logf(ssum);
      const int rg = row0 + row, a = rg >> 7, b = rg & 127;
      const float diag = S[row * 128 + b];
      if (cb != a) rowacc += (lse - diag);
    }
    __syncthreads();   // before next cb overwrites Qt/S/red
  }
  if (q == 0) rowsum[row] = rowacc;
  __syncthreads();
  if (tid == 0) {
    float tot = 0.f;
    for (int i = 0; i < 64; ++i) tot += rowsum[i];   // fixed order: deterministic
    partials[pbase + t * 16 + blk] = tot;
  }
}

__global__ void final_reduce(const float* __restrict__ partials, int n, float scale,
                             float* __restrict__ out) {
  if (threadIdx.x == 0 && blockIdx.x == 0) {
    float t = 0.f;
    for (int i = 0; i < n; ++i) t += partials[i];
    out[0] = t * scale;
  }
}

// ---------------------------------------------------------------------------
// Host launch
// ---------------------------------------------------------------------------
extern "C" void kernel_launch(void* const* d_in, const int* in_sizes, int n_in,
                              void* d_out, int out_size, void* d_ws, size_t ws_size,
                              hipStream_t stream) {
  (void)in_sizes; (void)n_in; (void)out_size; (void)ws_size;
  const float* states  = (const float*)d_in[0];   // (8,32,128,256)
  const float* futures = (const float*)d_in[1];   // (8,35,128,256)
  const float* actions = (const float*)d_in[2];   // (128,8,32,6)
  const float* Wp      = (const float*)d_in[3];   // (256,512)
  const float* bp      = (const float*)d_in[4];   // (512)
  const float* W1s[3]  = {(const float*)d_in[5],  (const float*)d_in[6],  (const float*)d_in[7]};
  const float* b1s[3]  = {(const float*)d_in[8],  (const float*)d_in[9],  (const float*)d_in[10]};
  const float* W2s[3]  = {(const float*)d_in[11], (const float*)d_in[12], (const float*)d_in[13]};
  const float* b2s[3]  = {(const float*)d_in[14], (const float*)d_in[15], (const float*)d_in[16]};
  float* outp = (float*)d_out;

  char* ws = (char*)d_ws;
  size_t off = 0;
  auto wsalloc = [&](size_t bytes) { char* p = ws + off; off += bytes; return p; };
  _Float16* states16  = (_Float16*)wsalloc((size_t)32768 * 256 * 2);
  _Float16* futures16 = (_Float16*)wsalloc((size_t)35840 * 256 * 2);
  _Float16* WpT       = (_Float16*)wsalloc((size_t)512 * 256 * 2);
  _Float16* W1T[3], *W2T[3];
  for (int i = 0; i < 3; ++i) W1T[i] = (_Float16*)wsalloc((size_t)512 * 512 * 2);
  for (int i = 0; i < 3; ++i) W2T[i] = (_Float16*)wsalloc((size_t)512 * 512 * 2);
  _Float16* proj = (_Float16*)wsalloc((size_t)32768 * 512 * 2);
  _Float16* tgt  = (_Float16*)wsalloc((size_t)35840 * 512 * 2);
  _Float16* hbuf = (_Float16*)wsalloc((size_t)32768 * 512 * 2);
  _Float16* pred = (_Float16*)wsalloc((size_t)32768 * 512 * 2);
  float* partials = (float*)wsalloc(1488 * 4);

  const dim3 blk(256);
  // One-time f16 staging of all GEMM operands
  cvt_f32_f16<<<dim3(8388608 / 2048), blk, 0, stream>>>(states, states16, 8388608);
  cvt_f32_f16<<<dim3(9175040 / 2048), blk, 0, stream>>>(futures, futures16, 9175040);
  transpose_w<<<dim3(131072 / 2048), blk, 0, stream>>>(Wp, WpT, 256);
  for (int i = 0; i < 3; ++i)
    transpose_w<<<dim3(262144 / 2048), blk, 0, stream>>>(W1s[i], W1T[i], 512);
  for (int i = 0; i < 3; ++i)
    transpose_w<<<dim3(262144 / 2048), blk, 0, stream>>>(W2s[i], W2T[i], 512);

  // proj = states @ Wp + bp ; tgt = l2norm(futures @ Wp + bp)
  gemm_plain<<<dim3(32768 / 64, 4), blk, 0, stream>>>(states16, WpT, bp, proj, 256);
  gemm_plain<<<dim3(35840 / 64, 4), blk, 0, stream>>>(futures16, WpT, bp, tgt, 256);
  normalize_rows<<<dim3(35840 / 8), blk, 0, stream>>>(tgt, 35840);

  const size_t dsmem = 132352;   // Pf 64KB + Qt 32KB + S 32KB + red/rowsum
  int pbase = 0;
  for (int kidx = 0; kidx < 3; ++kidx) {
    const int k = kidx + 1;
    const int tk = 32 - k + 1;
    const int M = 8 * tk * 128;   // 32768 / 31744 / 30720, all %64 == 0
    gemm_h<<<dim3(M / 64, 4), blk, 0, stream>>>(proj, W1T[kidx], W1s[kidx], b1s[kidx],
                                                actions, hbuf, tk, k);
    gemm_plain<<<dim3(M / 64, 4), blk, 0, stream>>>(hbuf, W2T[kidx], b2s[kidx], pred, 512);
    normalize_rows<<<dim3(M / 8), blk, 0, stream>>>(pred, M);
    logits_ce<<<dim3(tk, 16), blk, dsmem, stream>>>(pred, tgt, partials, tk, k, pbase);
    pbase += tk * 16;
  }
  // total / count, including the 1/B factor of the mean over b; count = 8*7*93
  final_reduce<<<1, 32, 0, stream>>>(partials, pbase, 1.0f / (128.0f * 5208.0f), outp);
}
